// WatershedFilter_33182917329004
// MI455X (gfx1250) — matrix operand order repaired and checked by reading, not compiled, not run on hardware
//
#include <hip/hip_runtime.h>
#include <hip/hip_bf16.h>

// ---------------- problem geometry ----------------
#define H 2048
#define W 2048
#define NPIX (H * W)
#define PAD 8
#define WP (W + 2 * PAD)      // 2064
#define NP (WP * WP)          // 4,260,096
#define INFC 1e30f

// temporal-blocked relaxation
#define TILE 64
#define KIT 8                 // sub-iterations per launch
#define LW (TILE + 2 * KIT)   // 80
#define RELAX_LAUNCHES 64     // 64 * 8 = 512 total iterations (== MAX_ITERS cap)

// ---------------- workspace layout (bytes) ----------------
#define O_GRAY   0ull
#define SZ_GRAY  ((unsigned long long)NP * 4ull)
#define O_SOBEL  (O_GRAY + SZ_GRAY)
#define SZ_SOBEL ((unsigned long long)NPIX * 4ull)
#define O_C12A   (O_SOBEL + SZ_SOBEL)
#define SZ_C12   ((unsigned long long)NP * 8ull)
#define O_C12B   (O_C12A + SZ_C12)
#define O_HIST1  (O_C12B + SZ_C12)
#define SZ_HIST1 (16384ull * 4ull)
#define O_H2A    (O_HIST1 + SZ_HIST1)
#define SZ_H2    (262144ull * 4ull)
#define O_H2B    (O_H2A + SZ_H2)
#define O_PARAMS (O_H2B + SZ_H2)

typedef _Float16 v16h __attribute__((ext_vector_type(16)));
typedef float    v8f  __attribute__((ext_vector_type(8)));
typedef unsigned int v4u __attribute__((ext_vector_type(4)));
typedef int          v8i __attribute__((ext_vector_type(8)));
typedef int          v4i __attribute__((ext_vector_type(4)));

// ---------------- TDM: 2D tile global -> LDS ----------------
// Builds the D# per CDNA5 ISA and issues tensor_load_to_lds.
__device__ __forceinline__ void tdm_load_2d(unsigned ldsAddr, const void* gaddr,
                                            unsigned dsCode,        // 2=4B, 3=8B
                                            unsigned td0, unsigned td1,
                                            unsigned tile0, unsigned tile1,
                                            unsigned stride /* data_size units */) {
  unsigned long long ga = (unsigned long long)gaddr;
  v4u g0;
  g0[0] = 1u;                                    // count=1 (valid descriptor)
  g0[1] = ldsAddr;                               // lds_addr (bytes)
  g0[2] = (unsigned)ga;                          // global_addr[31:0]
  g0[3] = (unsigned)((ga >> 32) & 0x01FFFFFFull) | (2u << 30);  // addr[56:32] | type=2
  v8i g1;
  g1[0] = (int)(dsCode << 16);                   // workgroup_mask=0, data_size
  g1[1] = (int)((td0 & 0xFFFFu) << 16);          // tensor_dim0[15:0]
  g1[2] = (int)((td0 >> 16) | ((td1 & 0xFFFFu) << 16));
  g1[3] = (int)((td1 >> 16) | (tile0 << 16));    // tile_dim0
  g1[4] = (int)(tile1 & 0xFFFFu);                // tile_dim1, tile_dim2=0
  g1[5] = (int)stride;                           // tensor_dim0_stride[31:0]
  g1[6] = 0;
  g1[7] = 0;
  v4i g2 = {};
  v4i g3 = {};
#if defined(__clang_major__) && (__clang_major__ >= 23)
  v8i z8 = {};
  __builtin_amdgcn_tensor_load_to_lds(g0, g1, g2, g3, z8, 0);
#else
  __builtin_amdgcn_tensor_load_to_lds(g0, g1, g2, g3, 0);
#endif
}

// ---------------- kernels ----------------
__global__ void k_zero(unsigned* __restrict__ p, int n) {
  int i = blockIdx.x * 256 + threadIdx.x;
  if (i < n) p[i] = 0u;
}

// gray with zero pad ring (zero pad == conv 'SAME' semantics for Sobel)
__global__ void k_gray(const float* __restrict__ img, float* __restrict__ grayP) {
  int i = blockIdx.x * 256 + threadIdx.x;
  if (i >= NP) return;
  int pr = i / WP, pc = i % WP;
  int r = pr - PAD, c = pc - PAD;
  float v = 0.f;
  if (r >= 0 && r < H && c >= 0 && c < W) {
    size_t o = (size_t)r * W + c;
    v = 0.2989f * img[o] + 0.587f * img[o + (size_t)NPIX] + 0.114f * img[o + 2 * (size_t)NPIX];
  }
  grayP[i] = v;
}

// Sobel magnitude via WMMA: vertical taps as per-lane adds on 3 row-shifted loads,
// horizontal taps as 16x16 tile (18-col window, zero-padded K=32) x banded matrix.
// One wave per 16x16 output tile; 8 tiles per 256-thread block.
__global__ void k_sobel_wmma(const float* __restrict__ grayP, float* __restrict__ sob) {
  int lane = threadIdx.x & 31;
  int wave = threadIdx.x >> 5;
  int tile = blockIdx.x * 8 + wave;
  int tr = tile >> 7, tc = tile & 127;            // 128x128 tiles
  int r0 = tr * 16, c0 = tc * 16;
  bool hi = lane >= 16;
  int m = lane & 15;                              // A: row M ; B/D: col N

  v16h a_s, a_d, bd, bs;
  const float* base = grayP + (size_t)(r0 + m - 1 + PAD) * WP + (size_t)(c0 - 1 + PAD);
#pragma unroll
  for (int e = 0; e < 16; ++e) {
    // A 16x32 f16 layout: lanes 0-15 hold K {0..7,16..23}; lanes 16-31 hold K {8..15,24..31}
    int col = hi ? (e < 8 ? 8 + e : 99) : (e < 8 ? e : e + 8);  // window col = K
    float vs = 0.f, vd = 0.f;
    if (col < 18) {                              // 18-wide window, rest zero
      float u  = base[col];
      float mm = base[col + WP];
      float d  = base[col + 2 * WP];
      vs = u + 2.f * mm + d;                     // vertical smooth [1,2,1]
      vd = d - u;                                // vertical diff  [-1,0,1]
    }
    a_s[e] = (_Float16)vs;
    a_d[e] = (_Float16)vd;
    // B 32x16 f16 layout (matches A): lanes 0-15 hold K {0..7,16..23} (col N=lane); lanes 16-31 hold K {8..15,24..31}
    int k = (e < 8) ? ((hi ? 8 : 0) + e) : ((hi ? 24 : 16) + (e - 8));
    int n = m;
    bd[e] = (_Float16)((k == n) ? -1.f : ((k == n + 2) ? 1.f : 0.f));                 // d/dx
    bs[e] = (_Float16)((k == n) ? 1.f : ((k == n + 1) ? 2.f : ((k == n + 2) ? 1.f : 0.f))); // smooth x
  }
  v8f zero = {};
  v8f gx = __builtin_amdgcn_wmma_f32_16x16x32_f16(false, a_s, false, bd, (short)0, zero, false, false);
  v8f gy = __builtin_amdgcn_wmma_f32_16x16x32_f16(false, a_d, false, bs, (short)0, zero, false, false);
#pragma unroll
  for (int v = 0; v < 8; ++v) {
    int M = v + (hi ? 8 : 0);                    // C/D layout: lane half selects M offset
    float g = sqrtf(gx[v] * gx[v] + gy[v] * gy[v]);
    sob[(size_t)(r0 + M) * W + (size_t)(c0 + m)] = g;
  }
}

__global__ void k_hist1(const float* __restrict__ sob, unsigned* __restrict__ h1) {
  int i = blockIdx.x * 256 + threadIdx.x;
  if (i < NPIX) atomicAdd(&h1[__float_as_uint(sob[i]) >> 18], 1u);
}

__global__ void k_select1(const unsigned* __restrict__ h1, unsigned* __restrict__ prm) {
  if (threadIdx.x != 0) return;
  const unsigned kLo = 4120903u;   // round(0.9825*(N-1))
  const unsigned kHi = 4162846u;   // round(0.9925*(N-1))
  unsigned cum = 0;
  for (int b = 0; b < 16384; ++b) {
    unsigned c = h1[b];
    if (kLo >= cum && kLo < cum + c) { prm[0] = (unsigned)b; prm[1] = kLo - cum; }
    if (kHi >= cum && kHi < cum + c) { prm[2] = (unsigned)b; prm[3] = kHi - cum; }
    cum += c;
  }
}

__global__ void k_hist2(const float* __restrict__ sob, const unsigned* __restrict__ prm,
                        unsigned* __restrict__ h2a, unsigned* __restrict__ h2b) {
  int i = blockIdx.x * 256 + threadIdx.x;
  if (i >= NPIX) return;
  unsigned bits = __float_as_uint(sob[i]);
  unsigned b = bits >> 18, low = bits & 0x3FFFFu;
  if (b == prm[0]) atomicAdd(&h2a[low], 1u);
  if (b == prm[2]) atomicAdd(&h2b[low], 1u);
}

__global__ void k_select2(const unsigned* __restrict__ h2a, const unsigned* __restrict__ h2b,
                          unsigned* __restrict__ prm) {
  int t = threadIdx.x;
  if (t >= 2) return;
  const unsigned* h = t ? h2b : h2a;
  unsigned bin = prm[t * 2], rank = prm[t * 2 + 1];
  unsigned cum = 0, sel = 0;
  for (int i = 0; i < 262144; ++i) {
    unsigned c = h[i];
    if (rank >= cum && rank < cum + c) sel = (unsigned)i;
    cum += c;
  }
  prm[4 + t] = (bin << 18) | sel;   // exact float bits of the order statistic
}

// seed both cost fields (interleaved float2), INF outside the image (padding ring)
__global__ void k_init(const float* __restrict__ grayP, const float* __restrict__ sob,
                       const unsigned* __restrict__ prm,
                       float2* __restrict__ ca, float2* __restrict__ cb) {
  int i = blockIdx.x * 256 + threadIdx.x;
  if (i >= NP) return;
  int pr = i / WP, pc = i % WP;
  int r = pr - PAD, c = pc - PAD;
  float2 v; v.x = INFC; v.y = INFC;
  if (r >= 0 && r < H && c >= 0 && c < W) {
    float s = sob[(size_t)r * W + c];
    float tLo = __uint_as_float(prm[4]);
    float tHi = __uint_as_float(prm[5]);
    float g = grayP[i];
    if (s <= tLo)      v.x = g;   // marker 1 (precedence, as in reference)
    else if (s >= tHi) v.y = g;   // marker 2
  }
  ca[i] = v;
  cb[i] = v;
}

// K=8 temporally-blocked relaxation on an 80x80 LDS halo tile, fed by TDM.
#define LDS_CBUF  (LW * LW * 8)                  // 51200 B (float2 tile)
#define LDS_TOTAL (2 * LDS_CBUF + LW * LW * 4)   // 128000 B
__global__ void k_relax(const float2* __restrict__ cin, float2* __restrict__ cout,
                        const float* __restrict__ grayP) {
  extern __shared__ char smem[];
  float2* bufA = (float2*)smem;
  float2* bufB = (float2*)(smem + LDS_CBUF);
  float*  gT   = (float*)(smem + 2 * LDS_CBUF);
  int tid = threadIdx.x;
  int orow = blockIdx.y * TILE;                  // tile origin in padded coords
  int ocol = blockIdx.x * TILE;

  if (tid == 0) {
    unsigned base = __builtin_amdgcn_groupstaticsize();  // dynamic-LDS base offset
    const void* gc = (const void*)(cin   + (size_t)orow * WP + ocol);
    const void* gg = (const void*)(grayP + (size_t)orow * WP + ocol);
    tdm_load_2d(base,                    gc, 3u, WP, WP, LW, LW, WP);  // 8B elems
    tdm_load_2d(base + 2u * LDS_CBUF,    gg, 2u, WP, WP, LW, LW, WP);  // 4B elems
    __builtin_amdgcn_s_wait_tensorcnt(0);
  }
  __syncthreads();

  // seed bufB's edge ring (relax loop never writes the outermost ring)
  for (int i = tid; i < LW; i += 256) {
    bufB[i] = bufA[i];
    bufB[(LW - 1) * LW + i] = bufA[(LW - 1) * LW + i];
    bufB[i * LW] = bufA[i * LW];
    bufB[i * LW + LW - 1] = bufA[i * LW + LW - 1];
  }
  __syncthreads();

  float2* cur = bufA;
  float2* nxt = bufB;
#pragma unroll 1
  for (int it = 0; it < KIT; ++it) {
    for (int i = tid; i < (LW - 2) * (LW - 2); i += 256) {
      int lr = 1 + i / (LW - 2);
      int lc = 1 + i % (LW - 2);
      float2 ctr = cur[lr * LW + lc];
      float2 nN = cur[(lr - 1) * LW + lc];
      float2 nS = cur[(lr + 1) * LW + lc];
      float2 nW = cur[lr * LW + lc - 1];
      float2 nE = cur[lr * LW + lc + 1];
      float mnx = fminf(fminf(nN.x, nS.x), fminf(nW.x, nE.x));
      float mny = fminf(fminf(nN.y, nS.y), fminf(nW.y, nE.y));
      float g = gT[lr * LW + lc];
      float2 res;
      res.x = fminf(ctr.x, fmaxf(mnx, g));
      res.y = fminf(ctr.y, fmaxf(mny, g));
      int gpr = orow + lr, gpc = ocol + lc;
      bool inside = (gpr >= PAD) && (gpr < PAD + H) && (gpc >= PAD) && (gpc < PAD + W);
      nxt[lr * LW + lc] = inside ? res : ctr;    // keep INF padding outside image
    }
    __syncthreads();
    float2* t = cur; cur = nxt; nxt = t;
  }

  // write exact 64x64 interior (cells >= KIT from tile edge are exact after KIT iters)
  for (int i = tid; i < TILE * TILE; i += 256) {
    int r = i >> 6, c = i & 63;
    cout[(size_t)(orow + PAD + r) * WP + (size_t)(ocol + PAD + c)] =
        cur[(KIT + r) * LW + (KIT + c)];
  }
}

__global__ void k_final(const float2* __restrict__ cfin, float* __restrict__ out) {
  int i = blockIdx.x * 256 + threadIdx.x;
  if (i >= NPIX) return;
  int r = i / W, c = i % W;
  float2 v = cfin[(size_t)(r + PAD) * WP + (size_t)(c + PAD)];
  out[i] = (v.y < v.x) ? 1.0f : 0.0f;            // (label - min) / max for labels {1,2}
}

// ---------------- host side ----------------
extern "C" void kernel_launch(void* const* d_in, const int* in_sizes, int n_in,
                              void* d_out, int out_size, void* d_ws, size_t ws_size,
                              hipStream_t stream) {
  (void)in_sizes; (void)n_in; (void)out_size; (void)ws_size;
  const float* img = (const float*)d_in[0];
  char* ws = (char*)d_ws;
  float*    grayP = (float*)(ws + O_GRAY);
  float*    sob   = (float*)(ws + O_SOBEL);
  float2*   ca    = (float2*)(ws + O_C12A);
  float2*   cb    = (float2*)(ws + O_C12B);
  unsigned* h1    = (unsigned*)(ws + O_HIST1);
  unsigned* h2a   = (unsigned*)(ws + O_H2A);
  unsigned* h2b   = (unsigned*)(ws + O_H2B);
  unsigned* prm   = (unsigned*)(ws + O_PARAMS);
  float*    out   = (float*)d_out;

  int nz = (int)((SZ_HIST1 + 2 * SZ_H2 + 256) / 4);   // hists + params are contiguous
  k_zero<<<(nz + 255) / 256, 256, 0, stream>>>(h1, nz);
  k_gray<<<(NP + 255) / 256, 256, 0, stream>>>(img, grayP);
  k_sobel_wmma<<<(128 * 128) / 8, 256, 0, stream>>>(grayP, sob);
  k_hist1<<<NPIX / 256, 256, 0, stream>>>(sob, h1);
  k_select1<<<1, 32, 0, stream>>>(h1, prm);
  k_hist2<<<NPIX / 256, 256, 0, stream>>>(sob, prm, h2a, h2b);
  k_select2<<<1, 32, 0, stream>>>(h2a, h2b, prm);
  k_init<<<(NP + 255) / 256, 256, 0, stream>>>(grayP, sob, prm, ca, cb);

  dim3 grid(W / TILE, H / TILE);
  float2* pin = ca;
  float2* pout = cb;
  for (int l = 0; l < RELAX_LAUNCHES; ++l) {
    k_relax<<<grid, 256, LDS_TOTAL, stream>>>(pin, pout, grayP);
    float2* t = pin; pin = pout; pout = t;
  }
  k_final<<<NPIX / 256, 256, 0, stream>>>(pin, out);
}
